// ArcFaceLoss_89867895701783
// MI455X (gfx1250) — matrix operand order repaired and checked by reading, not compiled, not run on hardware
//
#include <hip/hip_runtime.h>
#include <math.h>

// ---------------- problem constants ----------------
#define EMBED        512
#define NUM_CLASSES  100000
#define BATCH        512
#define ARC_SCALE    64.0f
#define COS_M        0.8775825618903728f   // cos(0.5)
#define SIN_M        0.4794255386042030f   // sin(0.5)
#define THRESH      -0.8775825618903728f   // cos(pi-0.5)
#define MARGIN_MULT  0.2397127693021015f   // sin(pi-0.5)*0.5
#define EPS_NORM     1e-12f

// ---------------- vector types ----------------
typedef __bf16        v16bf __attribute__((ext_vector_type(16)));
typedef float         v8f   __attribute__((ext_vector_type(8)));
typedef unsigned int  u32x4 __attribute__((ext_vector_type(4)));
typedef int           i32x4 __attribute__((ext_vector_type(4)));
typedef int           i32x8 __attribute__((ext_vector_type(8)));

#if __has_builtin(__builtin_amdgcn_tensor_load_to_lds) && \
    __has_builtin(__builtin_amdgcn_s_wait_tensorcnt)
#define USE_TDM 1
#else
#define USE_TDM 0
#endif

// round-to-nearest-even f32 -> bf16, packed x2 into a dword
static __device__ __forceinline__ unsigned pack_bf16x2(float lo, float hi) {
    union { float f; unsigned u; } a, b;
    a.f = lo; b.f = hi;
    unsigned ua = (a.u + 0x7FFFu + ((a.u >> 16) & 1u)) >> 16;
    unsigned ub = (b.u + 0x7FFFu + ((b.u >> 16) & 1u)) >> 16;
    return ua | (ub << 16);
}

// =====================================================================
// Kernel 1: L2-normalize rows of a [rows x 512] f32 matrix, emit bf16.
// One wave32 per row; each lane owns 4 float4 chunks (16 floats).
// =====================================================================
__global__ __launch_bounds__(256)
void arcface_rownorm_bf16(const float* __restrict__ src,
                          unsigned short* __restrict__ dst, // bf16 storage
                          int rows) {
    const int wave = threadIdx.x >> 5;
    const int lane = threadIdx.x & 31;
    const int row  = blockIdx.x * 8 + wave;
    if (row >= rows) return;

    const float4* p = (const float4*)(src + (size_t)row * EMBED);
    float4 v[4];
    float s = 0.f;
#pragma unroll
    for (int w = 0; w < 4; ++w) {
        v[w] = p[w * 32 + lane];
        s += v[w].x * v[w].x + v[w].y * v[w].y + v[w].z * v[w].z + v[w].w * v[w].w;
    }
#pragma unroll
    for (int off = 16; off >= 1; off >>= 1)
        s += __shfl_xor(s, off, 32);

    const float inv = 1.0f / fmaxf(sqrtf(s), EPS_NORM);

    uint2* dq = (uint2*)(dst + (size_t)row * EMBED);
#pragma unroll
    for (int w = 0; w < 4; ++w) {
        const int i = w * 32 + lane;
        uint2 o;
        o.x = pack_bf16x2(v[w].x * inv, v[w].y * inv);
        o.y = pack_bf16x2(v[w].z * inv, v[w].w * inv);
        dq[i] = o;
    }
}

// =====================================================================
// Kernel 2: cosine GEMM via v_wmma_f32_16x16x32_bf16 + fused ArcFace
// epilogue. TDM double-buffered B-tile staging (tensor_load_to_lds,
// TENSORcnt) when available; manual global->LDS copy fallback.
//   A (per wave)  : 16 classes x 32 K   (bf16 rows within a lane)
//   B (LDS tile)  : 32 K x 128 batch    (bf16)
//   D             : 16 classes x 16 batch, f32
// =====================================================================
#define CTILE 128
#define NTILE 128
#define KSTEP 32
#define EPAD  8          // LDS row pad -> 80B stride (16 dw data + 4 dw pad)

__global__ __launch_bounds__(256)
void arcface_wmma_gemm(const unsigned short* __restrict__ wbf,  // [C x 512] bf16
                       const unsigned short* __restrict__ ebf,  // [B x 512] bf16
                       const int* __restrict__ labels,          // [B]
                       float* __restrict__ rowSum,              // [B] exp-sum accum
                       float* __restrict__ tgt) {               // [B] target logit
    __shared__ unsigned short eT[2][NTILE][KSTEP + EPAD];
    __shared__ float lsum[NTILE];

    const int tid  = threadIdx.x;
    const int wave = tid >> 5;
    const int lane = tid & 31;

    if (tid < NTILE) lsum[tid] = 0.f;

    const int mBase = blockIdx.x * CTILE + wave * 16;   // class base for this wave
    const int nBase = blockIdx.y * NTILE;               // batch base for this WG

    // A-fragment addressing: lane<16 holds K {0..15}, lane>=16 {16..31}
    int arow = mBase + (lane & 15);
    if (arow >= NUM_CLASSES) arow = NUM_CLASSES - 1;    // clamp; masked in epilogue
    const size_t abase = (size_t)arow * EMBED + ((lane >> 4) ? 16 : 0);

#if USE_TDM
    // LDS byte offsets of the two tile buffers (flat LDS addr truncates to offset)
    const unsigned lds0 = (unsigned)(uintptr_t)&eT[0][0][0];
    const unsigned lds1 = (unsigned)(uintptr_t)&eT[1][0][0];
    const unsigned long long egbase =
        (unsigned long long)(uintptr_t)(ebf + (size_t)nBase * EMBED);

    // TDM D# descriptor: 2D tile 32(k) x 128(rows) bf16, row stride 512 elems,
    // LDS padding: 4 dwords after every 16 dwords (matches KSTEP+EPAD stride).
    auto issue_tile = [&](int k0, int buf) {
        if (wave != 0) return;
        const unsigned long long ga = egbase + (unsigned long long)k0 * 2ull;
        u32x4 g0;
        g0[0] = 1u;                                        // count=1, user mode
        g0[1] = buf ? lds1 : lds0;                         // lds_addr (bytes)
        g0[2] = (unsigned)(ga & 0xffffffffull);            // global_addr[31:0]
        g0[3] = (unsigned)((ga >> 32) & 0x01ffffffull)     // global_addr[56:32]
              | (2u << 30);                                // type=2 (image)
        i32x8 g1;
        g1[0] = (1 << 16)                                  // data_size = 2B
              | (1 << 20)                                  // pad_enable
              | (3 << 22)                                  // pad_interval: 16 dw
              | (3 << 25);                                 // pad_amount:   4 dw
        g1[1] = (int)(512u << 16);                         // tensor_dim0 = 512
        g1[2] = (int)(512u << 16);                         // tensor_dim1 = 512
        g1[3] = (int)(32u  << 16);                         // tile_dim0 = 32 (k)
        g1[4] = 128;                                       // tile_dim1 = 128 rows
        g1[5] = 512;                                       // tensor_dim0_stride
        g1[6] = 0;
        g1[7] = 0;
        i32x4 g2 = {0, 0, 0, 0};
        i32x4 g3 = {0, 0, 0, 0};
        i32x8 g4 = {0, 0, 0, 0, 0, 0, 0, 0};               // extra group (6-arg form)
        __builtin_amdgcn_tensor_load_to_lds(g0, g1, g2, g3, g4, 0);
    };
#else
    // manual staging: thread t copies 16 bf16 of row (t>>1), half (t&1)
    const int erow  = tid >> 1;
    const int ehalf = (tid & 1) * 16;
    const unsigned short* esrc = ebf + (size_t)(nBase + erow) * EMBED + ehalf;
    auto issue_tile = [&](int k0, int buf) {
        const uint4* s0 = (const uint4*)(esrc + k0);
        uint4* d0 = (uint4*)(&eT[buf][erow][ehalf]);
        d0[0] = s0[0];
        d0[1] = s0[1];
    };
#endif

    v8f acc[NTILE / 16] = {};   // 8 sub-tiles x 8 VGPRs

    issue_tile(0, 0);           // prologue: stage first tile

    for (int kk = 0; kk < EMBED / KSTEP; ++kk) {
        const int k0  = kk * KSTEP;
        const int cur = kk & 1;

        // ---- A fragment from global (bf16 weight stream) ----
        union { uint4 q[2]; v16bf v; } A;
        A.q[0] = *(const uint4*)(wbf + abase + k0);
        A.q[1] = *(const uint4*)(wbf + abase + k0 + 8);
        __builtin_prefetch(wbf + abase + k0 + KSTEP, 0, 1);

#if USE_TDM
        if (wave == 0) __builtin_amdgcn_s_wait_tensorcnt(0);
#endif
        __syncthreads();   // tile `cur` ready; previous tile's readers done

        // overlap: DMA/copy next tile into the other buffer during compute
        if (kk + 1 < EMBED / KSTEP) issue_tile(k0 + KSTEP, cur ^ 1);

        // ---- preload all 8 B fragments, then 8 back-to-back WMMAs ----
        uint4 bq[NTILE / 16][2];
#pragma unroll
        for (int nt = 0; nt < NTILE / 16; ++nt) {
            const int brow = nt * 16 + (lane & 15);      // batch column
            const int bcol = (lane >> 4) * 16;           // K half
            const uint4* sp = (const uint4*)(&eT[cur][brow][bcol]);
            bq[nt][0] = sp[0];
            bq[nt][1] = sp[1];
        }
#pragma unroll
        for (int nt = 0; nt < NTILE / 16; ++nt) {
            union { uint4 q[2]; v16bf v; } Bf;
            Bf.q[0] = bq[nt][0];
            Bf.q[1] = bq[nt][1];
            acc[nt] = __builtin_amdgcn_wmma_f32_16x16x32_bf16(
                false, A.v, false, Bf.v, (short)0, acc[nt], false, false);
        }
    }

    // ---- fused ArcFace epilogue ----
    // D layout: lane<16 -> col n=lane, rows mBase+0..7 ; lane>=16 -> col lane-16, rows +8..15
    const int mOff = (lane >> 4) * 8;
#pragma unroll
    for (int nt = 0; nt < NTILE / 16; ++nt) {
        const int b = nBase + nt * 16 + (lane & 15);
        const int lab = labels[b];
        float partial = 0.f;
#pragma unroll
        for (int r = 0; r < 8; ++r) {
            const int cls = mBase + mOff + r;
            if (cls < NUM_CLASSES) {
                const float c = acc[nt][r];
                float logit;
                if (cls == lab) {
                    const float sine = sqrtf(fmaxf(1.0f - c * c, 0.0f));
                    float phi = c * COS_M - sine * SIN_M;
                    phi = (c > THRESH) ? phi : (c - MARGIN_MULT);
                    logit = ARC_SCALE * phi;
                    tgt[b] = logit;                      // exactly one writer per b
                } else {
                    logit = ARC_SCALE * c;
                }
                partial += __expf(logit - ARC_SCALE);    // fixed shift = max logit
            }
        }
        partial += __shfl_xor(partial, 16, 32);          // join the two class halves
        if (lane < 16)
            atomicAdd(&lsum[nt * 16 + lane], partial);   // ds_add_f32
    }

    __syncthreads();
    if (tid < NTILE)
        atomicAdd(&rowSum[nBase + tid], lsum[tid]);      // 8x fewer global atomics
}

// =====================================================================
// Kernel 3: nll_b = log(rowSum[b]) + 64 - tgt[b];  out = mean(nll)
// =====================================================================
__global__ __launch_bounds__(256)
void arcface_finalize(const float* __restrict__ rowSum,
                      const float* __restrict__ tgt,
                      float* __restrict__ out) {
    __shared__ float wsum[8];
    const int tid = threadIdx.x;
    float a = 0.f;
#pragma unroll
    for (int i = 0; i < 2; ++i) {
        const int b = tid + i * 256;
        a += (logf(rowSum[b]) + ARC_SCALE) - tgt[b];
    }
#pragma unroll
    for (int off = 16; off >= 1; off >>= 1)
        a += __shfl_xor(a, off, 32);
    if ((tid & 31) == 0) wsum[tid >> 5] = a;
    __syncthreads();
    if (tid == 0) {
        float t = 0.f;
#pragma unroll
        for (int w = 0; w < 8; ++w) t += wsum[w];
        out[0] = t / (float)BATCH;
    }
}

// =====================================================================
// Host-side launch
// =====================================================================
extern "C" void kernel_launch(void* const* d_in, const int* in_sizes, int n_in,
                              void* d_out, int out_size, void* d_ws, size_t ws_size,
                              hipStream_t stream) {
    const float* d_emb    = (const float*)d_in[0];   // [512 x 512] f32
    const int*   d_labels = (const int*)d_in[1];     // [512] int32
    const float* d_weight = (const float*)d_in[2];   // [100000 x 512] f32

    // workspace layout
    char* ws = (char*)d_ws;
    unsigned short* d_wbf = (unsigned short*)ws;                 // 102.4 MB
    ws += (size_t)NUM_CLASSES * EMBED * sizeof(unsigned short);
    unsigned short* d_ebf = (unsigned short*)ws;                 // 512 KB
    ws += (size_t)BATCH * EMBED * sizeof(unsigned short);
    float* d_rowSum = (float*)ws;  ws += BATCH * sizeof(float);
    float* d_tgt    = (float*)ws;  ws += BATCH * sizeof(float);

    (void)hipMemsetAsync(d_rowSum, 0, BATCH * sizeof(float), stream);

    // normalize + convert to bf16 (one wave per row)
    arcface_rownorm_bf16<<<BATCH / 8, 256, 0, stream>>>(d_emb, d_ebf, BATCH);
    arcface_rownorm_bf16<<<(NUM_CLASSES + 7) / 8, 256, 0, stream>>>(d_weight, d_wbf, NUM_CLASSES);

    // WMMA cosine GEMM + fused margin / exp-sum epilogue
    dim3 grid((NUM_CLASSES + CTILE - 1) / CTILE, BATCH / NTILE);
    arcface_wmma_gemm<<<grid, 256, 0, stream>>>(d_wbf, d_ebf, d_labels, d_rowSum, d_tgt);

    // final scalar reduction
    arcface_finalize<<<1, 256, 0, stream>>>(d_rowSum, d_tgt, (float*)d_out);
}
